// MaskRCNN_41970420416863
// MI455X (gfx1250) — matrix operand order repaired and checked by reading, not compiled, not run on hardware
//
#include <hip/hip_runtime.h>
#include <hip/hip_bf16.h>

#define N_PRE   2000
#define N_POST  1000
#define MASKW   64            // u32 words of pitch per NMS mask row
#define NW      63            // ceil(2000/32)
#define NCHN    256

typedef __attribute__((ext_vector_type(16))) __bf16       v16bf;
typedef __attribute__((ext_vector_type(8)))  float        v8f;
typedef __attribute__((ext_vector_type(8)))  unsigned int v8u;

// ---------- helpers ----------
__device__ __forceinline__ unsigned f2bf(float f) {
  unsigned u = __float_as_uint(f);
  return (u + 0x7FFFu + ((u >> 16) & 1u)) >> 16;   // round-to-nearest-even bf16
}
__device__ __forceinline__ unsigned pk2(float lo, float hi) {
  return f2bf(lo) | (f2bf(hi) << 16);
}
__device__ __forceinline__ bool prec(unsigned ka, unsigned ia, unsigned kb, unsigned ib) {
  // strict "comes first": larger score first, ties -> smaller original index (jax top_k order)
  return (ka > kb) || (ka == kb && ia < ib);
}

// ---------- 1. decode boxes + sortable score keys ----------
__global__ void k_decode(const float* __restrict__ rc, const float* __restrict__ rb,
                         const float* __restrict__ anch, float* __restrict__ boxes,
                         unsigned* __restrict__ keys, int A) {
  int a = blockIdx.x * blockDim.x + threadIdx.x;
  if (a >= A) return;
  float sc = rc[a * 2 + 1];
  float d0 = rb[a * 4 + 0] * 0.1f, d1 = rb[a * 4 + 1] * 0.1f;
  float d2 = rb[a * 4 + 2] * 0.2f, d3 = rb[a * 4 + 3] * 0.2f;
  float ay1 = anch[a * 4 + 0], ax1 = anch[a * 4 + 1];
  float ay2 = anch[a * 4 + 2], ax2 = anch[a * 4 + 3];
  float h = ay2 - ay1, w = ax2 - ax1;
  float cy = ay1 + 0.5f * h + d0 * h;
  float cx = ax1 + 0.5f * w + d1 * w;
  h *= expf(d2);  w *= expf(d3);
  float y1 = cy - 0.5f * h, x1 = cx - 0.5f * w;
  float y2 = y1 + h,        x2 = x1 + w;
  y1 = fminf(fmaxf(y1, 0.f), 1024.f);  x1 = fminf(fmaxf(x1, 0.f), 1024.f);
  y2 = fminf(fmaxf(y2, 0.f), 1024.f);  x2 = fminf(fmaxf(x2, 0.f), 1024.f);
  boxes[a * 4 + 0] = y1; boxes[a * 4 + 1] = x1;
  boxes[a * 4 + 2] = y2; boxes[a * 4 + 3] = x2;
  unsigned ub = __float_as_uint(sc);
  keys[a] = (ub & 0x80000000u) ? ~ub : (ub | 0x80000000u);   // monotone float->u32
}

// ---------- 2. radix-select threshold for top-2000 ----------
__global__ void k_init(unsigned* __restrict__ state, unsigned* __restrict__ hist) {
  if (threadIdx.x < 256) hist[threadIdx.x] = 0u;
  if (threadIdx.x == 0) { state[0] = 0u; state[1] = N_PRE; state[2] = 0u; }
}

__global__ void k_hist(const unsigned* __restrict__ keys, const unsigned* __restrict__ state,
                       unsigned* __restrict__ hist, int shift, int A) {
  __shared__ unsigned lh[256];
  lh[threadIdx.x] = 0u;
  __syncthreads();
  int a = blockIdx.x * 256 + threadIdx.x;
  if (a < A) {
    unsigned key = keys[a], prefix = state[0];
    unsigned maskHi = (shift == 24) ? 0u : (0xFFFFFFFFu << (shift + 8));
    if (((key ^ prefix) & maskHi) == 0u)
      atomicAdd(&lh[(key >> shift) & 255u], 1u);
  }
  __syncthreads();
  if (lh[threadIdx.x]) atomicAdd(&hist[threadIdx.x], lh[threadIdx.x]);
}

__global__ void k_select(unsigned* __restrict__ hist, unsigned* __restrict__ state, int shift) {
  unsigned k = state[1], prefix = state[0], cum = 0u;
  for (int d = 255; d >= 0; --d) {
    unsigned c = hist[d];
    if (cum + c >= k) { prefix |= ((unsigned)d) << shift; k -= cum; break; }
    cum += c;
  }
  state[0] = prefix; state[1] = k;
  if (shift == 0) state[2] = N_PRE - k;     // count(key > threshold)
  for (int d = 0; d < 256; ++d) hist[d] = 0u;
}

// ---------- 3. stable compaction of the 2000 selected ----------
__global__ void k_count(const unsigned* __restrict__ keys, const unsigned* __restrict__ state,
                        unsigned* __restrict__ cnta, unsigned* __restrict__ cntt, int A) {
  __shared__ unsigned sa, st;
  if (threadIdx.x == 0) { sa = 0u; st = 0u; }
  __syncthreads();
  int a = blockIdx.x * 256 + threadIdx.x;
  if (a < A) {
    unsigned key = keys[a], T = state[0];
    if (key > T) atomicAdd(&sa, 1u);
    else if (key == T) atomicAdd(&st, 1u);
  }
  __syncthreads();
  if (threadIdx.x == 0) { cnta[blockIdx.x] = sa; cntt[blockIdx.x] = st; }
}

__global__ void k_scanoffs(const unsigned* __restrict__ cnta, const unsigned* __restrict__ cntt,
                           unsigned* __restrict__ offa, unsigned* __restrict__ offt, int NB) {
  unsigned ra = 0u, rt = 0u;
  for (int b = 0; b < NB; ++b) {
    offa[b] = ra; ra += cnta[b];
    offt[b] = rt; rt += cntt[b];
  }
}

__global__ void k_compact(const unsigned* __restrict__ keys, const unsigned* __restrict__ state,
                          const unsigned* __restrict__ offa, const unsigned* __restrict__ offt,
                          unsigned* __restrict__ selkey, unsigned* __restrict__ selidx, int A) {
  int b = blockIdx.x;
  unsigned T = state[0], tot_a = state[2];
  unsigned pa = offa[b], pt = offt[b];
  int base = b * 256;
  for (int i = 0; i < 256; ++i) {
    int a = base + i;
    if (a >= A) break;
    unsigned key = keys[a];
    if (key > T) {
      unsigned slot = pa++;
      selkey[slot] = key; selidx[slot] = (unsigned)a;
    } else if (key == T) {
      unsigned slot = tot_a + pt++;
      if (slot < N_PRE) { selkey[slot] = key; selidx[slot] = (unsigned)a; }
    }
  }
}

// ---------- 4. bitonic sort of the 2000 (desc score, asc index) ----------
__global__ void k_sort(const unsigned* __restrict__ selkey, const unsigned* __restrict__ selidx,
                       const float* __restrict__ boxes, float* __restrict__ sboxes) {
  __shared__ unsigned sk[2048], si[2048];
  for (unsigned t = threadIdx.x; t < 2048; t += 1024) {
    if (t < N_PRE) { sk[t] = selkey[t]; si[t] = selidx[t]; }
    else           { sk[t] = 0u;        si[t] = 0xFFFFFFFFu; }
  }
  for (unsigned sz = 2; sz <= 2048; sz <<= 1) {
    for (unsigned st = sz >> 1; st > 0; st >>= 1) {
      __syncthreads();
      for (unsigned t = threadIdx.x; t < 2048; t += 1024) {
        unsigned j = t ^ st;
        if (j > t) {
          bool up = ((t & sz) == 0u);
          unsigned ka = sk[t], ia = si[t], kb = sk[j], ib = si[j];
          bool sw = up ? prec(kb, ib, ka, ia) : prec(ka, ia, kb, ib);
          if (sw) { sk[t] = kb; si[t] = ib; sk[j] = ka; si[j] = ia; }
        }
      }
    }
  }
  __syncthreads();
  for (unsigned t = threadIdx.x; t < N_PRE; t += 1024) {
    unsigned s = si[t];
    sboxes[t * 4 + 0] = boxes[s * 4 + 0];
    sboxes[t * 4 + 1] = boxes[s * 4 + 1];
    sboxes[t * 4 + 2] = boxes[s * 4 + 2];
    sboxes[t * 4 + 3] = boxes[s * 4 + 3];
  }
}

// ---------- 5. NMS: IoU bitmask (one wave32 per 32-col strip) ----------
__global__ void k_mask(const float* __restrict__ b, unsigned* __restrict__ mask) {
  int i = blockIdx.y;
  int j = blockIdx.x * 32 + threadIdx.x;
  __shared__ unsigned wrd;
  if (threadIdx.x == 0) wrd = 0u;
  __syncthreads();
  bool bit = false;
  if (j < N_PRE && j > i) {
    float iy1 = b[i*4+0], ix1 = b[i*4+1], iy2 = b[i*4+2], ix2 = b[i*4+3];
    float jy1 = b[j*4+0], jx1 = b[j*4+1], jy2 = b[j*4+2], jx2 = b[j*4+3];
    float ai = (iy2 - iy1 + 1.f) * (ix2 - ix1 + 1.f);
    float aj = (jy2 - jy1 + 1.f) * (jx2 - jx1 + 1.f);
    float yy1 = fmaxf(iy1, jy1), xx1 = fmaxf(ix1, jx1);
    float yy2 = fminf(iy2, jy2), xx2 = fminf(ix2, jx2);
    float inter = fmaxf(yy2 - yy1 + 1.f, 0.f) * fmaxf(xx2 - xx1 + 1.f, 0.f);
    float iou = inter / (ai + aj - inter);
    bit = iou > 0.7f;
  }
  if (bit) atomicOr(&wrd, 1u << threadIdx.x);
  __syncthreads();
  if (threadIdx.x == 0) mask[(size_t)i * MASKW + blockIdx.x] = wrd;
}

// ---------- 6. sequential greedy NMS scan ----------
__global__ void k_nmsscan(const unsigned* __restrict__ mask, unsigned* __restrict__ keepw) {
  __shared__ unsigned remv[NW];
  int t = threadIdx.x;
  if (t < NW) remv[t] = 0u;
  __syncthreads();
  for (int i = 0; i < N_PRE; ++i) {
    unsigned w = remv[i >> 5];
    bool kept = ((w >> (i & 31)) & 1u) == 0u;
    __syncthreads();
    if (kept && t < NW) remv[t] |= mask[(size_t)i * MASKW + t];
    __syncthreads();
  }
  if (t < NW) {
    unsigned kw = ~remv[t];
    if (t == NW - 1) kw &= 0xFFFFu;     // 2000 = 62*32 + 16
    keepw[t] = kw;
  }
}

// ---------- 7. gather <=1000 kept rois, level assignment, normalize ----------
__global__ void k_rois(const unsigned* __restrict__ keepw, const float* __restrict__ sboxes,
                       float* __restrict__ roisn, int* __restrict__ lvl) {
  __shared__ int sel[N_POST];
  __shared__ int snk;
  if (threadIdx.x == 0) {
    int r = 0;
    for (int i = 0; i < N_PRE; ++i)
      if ((keepw[i >> 5] >> (i & 31)) & 1u) { if (r < N_POST) sel[r] = i; r++; }
    snk = (r < N_POST) ? r : N_POST;
  }
  __syncthreads();
  for (int r = threadIdx.x; r < N_POST; r += blockDim.x) {
    float b0 = 0.f, b1 = 0.f, b2 = 0.f, b3 = 0.f;
    if (r < snk) {
      int i = sel[r];
      b0 = sboxes[i*4+0]; b1 = sboxes[i*4+1]; b2 = sboxes[i*4+2]; b3 = sboxes[i*4+3];
    }
    float h = b2 - b0, w = b3 - b1;
    float l = rintf(log2f(sqrtf(fmaxf(h * w, 1e-6f)) / 224.f)) + 4.f;  // ties-to-even like jnp.round
    int lv = (int)l; lv = lv < 2 ? 2 : (lv > 5 ? 5 : lv);
    const float inv = 1.f / 1024.f;
    roisn[r*4+0] = b0 * inv; roisn[r*4+1] = b1 * inv;
    roisn[r*4+2] = b2 * inv; roisn[r*4+3] = b3 * inv;
    lvl[r] = lv;
  }
}

// ---------- 8. ROIAlign via block-diagonal WMMA ----------
// out[pos,ch] = sum_t W[pos,t] * tap[pos,t,ch];  M=16 positions, K=32 (=8 pos x 4 taps,
// two K-passes per group), N=16 channels, f32 accumulate in the C/D fragment.
// Padded positions (49..63) are clamped to position 48's tap addresses with zero weights
// in A, so every gather is unconditional.  Per channel tile, all 32 taps are loaded into
// registers FIRST (one long clause of independent global_load_b32), then packed to bf16
// and fed to the two WMMAs.  Output uses non-temporal stores so the streamed 50 MB result
// does not evict the L2-resident feature pyramids.
__global__ void k_roialign(const float* __restrict__ p2, const float* __restrict__ p3,
                           const float* __restrict__ p4, const float* __restrict__ p5,
                           const float* __restrict__ roisn, const int* __restrict__ lvl,
                           float* __restrict__ out) {
  __shared__ int   stap[64][4];   // tap offsets within one channel plane (y*H+x)
  __shared__ float swt[64][4];    // bilinear weights (zero for padded positions)
  int roi = blockIdx.x;
  int lv = lvl[roi];
  int H = 256 >> (lv - 2);
  const float* f = (lv == 2) ? p2 : (lv == 3) ? p3 : (lv == 4) ? p4 : p5;
  int tid = threadIdx.x;
  if (tid < 64) {
    int pos = (tid < 49) ? tid : 48;            // clamp padded slots
    int py = pos / 7, px = pos - py * 7;
    float by1 = roisn[roi*4+0], bx1 = roisn[roi*4+1];
    float by2 = roisn[roi*4+2], bx2 = roisn[roi*4+3];
    float hm1 = (float)(H - 1);
    float ys = by1 * hm1 + (float)py * ((by2 - by1) * hm1 / 6.0f);
    float y0f = fminf(fmaxf(floorf(ys), 0.f), hm1);
    int   y0  = (int)y0f;
    int   y1i = min(y0 + 1, H - 1);
    float wy  = fminf(fmaxf(ys, 0.f), hm1) - y0f;
    float xs = bx1 * hm1 + (float)px * ((bx2 - bx1) * hm1 / 6.0f);
    float x0f = fminf(fmaxf(floorf(xs), 0.f), hm1);
    int   x0  = (int)x0f;
    int   x1i = min(x0 + 1, H - 1);
    float wx  = fminf(fmaxf(xs, 0.f), hm1) - x0f;
    stap[tid][0] = y0  * H + x0;
    stap[tid][1] = y0  * H + x1i;
    stap[tid][2] = y1i * H + x0;
    stap[tid][3] = y1i * H + x1i;
    float z = (tid < 49) ? 1.f : 0.f;
    swt[tid][0] = (1.f - wy) * (1.f - wx) * z;
    swt[tid][1] = (1.f - wy) * wx         * z;
    swt[tid][2] = wy         * (1.f - wx) * z;
    swt[tid][3] = wy         * wx         * z;
  }
  __syncthreads();
  int lane = tid & 31, wv = tid >> 5;
  size_t HW = (size_t)H * H;
#pragma unroll
  for (int g4 = 0; g4 < 4; ++g4) {              // position groups 0..63 (49 used)
    // ---- A fragments (block-diagonal bilinear weights), shared by all channel tiles
    v16bf afr[2];
    {
      int m = lane & 15;
#pragma unroll
      for (int p = 0; p < 2; ++p) {
        v8u au;
#pragma unroll
        for (int v = 0; v < 8; ++v) {
          float wgt[2];
#pragma unroll
          for (int h = 0; h < 2; ++h) {
            int K = ((v >= 4) ? 16 : 0) + ((lane >= 16) ? 8 : 0) + ((v & 3) << 1) + h;
            int q = K >> 2, t = K & 3;
            float wraw = swt[16 * g4 + m][t];           // unconditional LDS read
            wgt[h] = (m == p * 8 + q) ? wraw : 0.f;     // v_cndmask, no branch
          }
          au[v] = pk2(wgt[0], wgt[1]);
        }
        afr[p] = __builtin_bit_cast(v16bf, au);
      }
    }
    // ---- tap offsets for both K-passes, hoisted out of the channel-tile loop
    int toff[2][8][2];
#pragma unroll
    for (int p = 0; p < 2; ++p)
#pragma unroll
      for (int v = 0; v < 8; ++v)
#pragma unroll
        for (int h = 0; h < 2; ++h) {
          int K = ((v >= 4) ? 16 : 0) + ((lane >= 16) ? 8 : 0) + ((v & 3) << 1) + h;
          int q = K >> 2, t = K & 3;
          int pos = 16 * g4 + 8 * p + q;                // always a valid table slot
          toff[p][v][h] = stap[pos][t];
        }
    // ---- two channel tiles per wave: 8 waves x 2 = 16 tiles = 256 channels
#pragma unroll
    for (int ni = 0; ni < 2; ++ni) {
      int ch = (wv + 8 * ni) * 16 + (lane & 15);
      const float* fc = f + (size_t)ch * HW;
      // phase 1: issue ALL 32 independent gathers (one clause, one wait)
      float tv[2][16];
#pragma unroll
      for (int p = 0; p < 2; ++p)
#pragma unroll
        for (int v = 0; v < 8; ++v) {
          tv[p][2 * v + 0] = fc[toff[p][v][0]];
          tv[p][2 * v + 1] = fc[toff[p][v][1]];
        }
      // phase 2: pack to bf16 and run the two K-passes
      v8f acc = {0.f, 0.f, 0.f, 0.f, 0.f, 0.f, 0.f, 0.f};
#pragma unroll
      for (int p = 0; p < 2; ++p) {
        v8u bu;
#pragma unroll
        for (int v = 0; v < 8; ++v)
          bu[v] = pk2(tv[p][2 * v + 0], tv[p][2 * v + 1]);
        v16bf bfr = __builtin_bit_cast(v16bf, bu);
        acc = __builtin_amdgcn_wmma_f32_16x16x32_bf16(false, afr[p], false, bfr,
                                                      (short)0, acc, false, false);
      }
      float* op = out + ((size_t)roi * NCHN + ch) * 49;
#pragma unroll
      for (int r = 0; r < 8; ++r) {
        int mm = (lane < 16) ? r : (r + 8);
        int pos = 16 * g4 + mm;
        if (pos < 49)
          __builtin_nontemporal_store(acc[r], &op[pos]);   // NT: don't evict fmaps from L2
      }
    }
  }
}

// ---------- launch ----------
extern "C" void kernel_launch(void* const* d_in, const int* in_sizes, int n_in,
                              void* d_out, int out_size, void* d_ws, size_t ws_size,
                              hipStream_t stream) {
  const float* rpn_class = (const float*)d_in[0];
  const float* rpn_bbox  = (const float*)d_in[1];
  const float* p2        = (const float*)d_in[2];
  const float* p3        = (const float*)d_in[3];
  const float* p4        = (const float*)d_in[4];
  const float* p5        = (const float*)d_in[5];
  const float* anchors   = (const float*)d_in[6];
  float* outp = (float*)d_out;

  const int A  = in_sizes[0] / 2;          // 261888
  const int NB = (A + 255) / 256;          // 1023

  char* w = (char*)d_ws;
  size_t o = 0;
  auto take = [&](size_t n) { void* p = w + o; o += (n + 255) & ~(size_t)255; return p; };
  float*    boxes  = (float*)    take((size_t)A * 16);
  unsigned* keys   = (unsigned*) take((size_t)A * 4);
  unsigned* hist   = (unsigned*) take(1024);
  unsigned* state  = (unsigned*) take(64);
  unsigned* cnta   = (unsigned*) take((size_t)NB * 4);
  unsigned* cntt   = (unsigned*) take((size_t)NB * 4);
  unsigned* offa   = (unsigned*) take((size_t)NB * 4);
  unsigned* offt   = (unsigned*) take((size_t)NB * 4);
  unsigned* selkey = (unsigned*) take(2048 * 4);
  unsigned* selidx = (unsigned*) take(2048 * 4);
  float*    sboxes = (float*)    take((size_t)N_PRE * 16);
  unsigned* maskb  = (unsigned*) take((size_t)N_PRE * MASKW * 4);
  unsigned* keepw  = (unsigned*) take((NW + 1) * 4);
  float*    roisn  = (float*)    take((size_t)N_POST * 16);
  int*      lvl    = (int*)      take((size_t)N_POST * 4);
  (void)ws_size; (void)n_in; (void)out_size;

  k_init<<<1, 256, 0, stream>>>(state, hist);
  k_decode<<<NB, 256, 0, stream>>>(rpn_class, rpn_bbox, anchors, boxes, keys, A);

  const int shifts[4] = {24, 16, 8, 0};
  for (int pshift = 0; pshift < 4; ++pshift) {
    k_hist<<<NB, 256, 0, stream>>>(keys, state, hist, shifts[pshift], A);
    k_select<<<1, 1, 0, stream>>>(hist, state, shifts[pshift]);
  }

  k_count<<<NB, 256, 0, stream>>>(keys, state, cnta, cntt, A);
  k_scanoffs<<<1, 1, 0, stream>>>(cnta, cntt, offa, offt, NB);
  k_compact<<<NB, 1, 0, stream>>>(keys, state, offa, offt, selkey, selidx, A);

  k_sort<<<1, 1024, 0, stream>>>(selkey, selidx, boxes, sboxes);

  k_mask<<<dim3(NW, N_PRE), 32, 0, stream>>>(sboxes, maskb);
  k_nmsscan<<<1, 64, 0, stream>>>(maskb, keepw);

  k_rois<<<1, 256, 0, stream>>>(keepw, sboxes, roisn, lvl);

  k_roialign<<<N_POST, 256, 0, stream>>>(p2, p3, p4, p5, roisn, lvl, outp);
}
